// TransformerModel_86973087744334
// MI455X (gfx1250) — matrix operand order repaired and checked by reading, not compiled, not run on hardware
//
#include <hip/hip_runtime.h>
#include <math.h>

// ---------------- problem constants (match reference) ----------------
static constexpr int Bc  = 4;
static constexpr int Lc  = 512;
static constexpr int Vc  = 32000;
static constexpr int Dc  = 1024;
static constexpr int Hc  = 16;
static constexpr int Fc  = 4096;
static constexpr int NLc = 6;
static constexpr int DHc = 64;
static constexpr int Mc  = Bc * Lc;   // 2048 rows of activations

// ---------------- vector types for WMMA ----------------
typedef __attribute__((ext_vector_type(16))) __bf16         bf16x16;
typedef __attribute__((ext_vector_type(8)))  float          f32x8;
typedef __attribute__((ext_vector_type(8)))  unsigned short u16x8;
typedef __attribute__((ext_vector_type(4)))  unsigned short u16x4;

union Frag { bf16x16 bf; u16x8 h[2]; };

__device__ __forceinline__ unsigned short f2bf(float f) {
  // f32 -> bf16 round-to-nearest-even
  union { float f; unsigned u; } v; v.f = f;
  unsigned r = v.u + 0x7FFFu + ((v.u >> 16) & 1u);
  return (unsigned short)(r >> 16);
}

// =====================================================================
// Double-buffered bf16 WMMA GEMM:  C = epi(alpha * A @ B)
//   A : bf16 [M,K] row-major  -> GLOBAL_LOAD_ASYNC_TO_LDS_B128 copy
//   B : BF16B ? bf16 [N,K] row-major (async DMA copy)
//             : f32  [K,N] row-major (transpose-convert; A.V path)
//   C : OUTBF ? bf16 : f32
// Pipeline: issue async DMA for tile k+1, WMMA tile k, s_wait_asynccnt,
// one barrier per iteration. Batched via grid.z with (batch/Hdiv,batch%Hdiv).
// =====================================================================
template<int BM, int BN, int WROWS, int WCOLS, bool BF16B, bool OUTBF,
         bool GELU, bool HASBIAS, bool HASRES, bool HASADD>
__global__ __launch_bounds__(256) void gemm_wmma(
    const unsigned short* __restrict__ Ag, const void* __restrict__ Bg,
    void* __restrict__ Cg,
    const float* __restrict__ bias, const float* __restrict__ res,
    const float* __restrict__ add,
    int K, int lda, int ldb, int ldc, int ldadd,
    long long soA, long long siA, long long soB, long long siB,
    long long soC, long long siC, long long soAdd, int Hdiv, float alpha)
{
  constexpr int BK  = 32;
  constexpr int WTM = BM / WROWS, WTN = BN / WCOLS;
  constexpr int FM  = WTM / 16,  FN  = WTN / 16;

  __shared__ unsigned short As[2][BM][BK];   // [m][k] bf16 bits
  __shared__ unsigned short Bs[2][BN][BK];   // [n][k] bf16 bits (N-major)

  const int tid  = threadIdx.x;
  const int lane = tid & 31;
  const int wave = tid >> 5;
  const int wr   = wave / WCOLS, wc = wave % WCOLS;

  const int batch = blockIdx.z;
  const long long bo = batch / Hdiv, bi = batch % Hdiv;
  const unsigned short* A  = Ag + bo * soA + bi * siA;
  const unsigned short* BH = (const unsigned short*)Bg + bo * soB + bi * siB;
  const float*          BF = (const float*)Bg + bo * soB + bi * siB;
  float*          CF = (float*)Cg + bo * soC + bi * siC;
  unsigned short* CH = (unsigned short*)Cg + bo * soC + bi * siC;
  const float* R  = HASRES ? (res + bo * soC + bi * siC) : nullptr;
  const float* Ad = HASADD ? (add + bo * soAdd) : nullptr;

  const int tileM = blockIdx.y * BM;
  const int tileN = blockIdx.x * BN;

  // wave-relative LDS byte offsets for the async-DMA destinations
  const unsigned AsBase = (unsigned)(unsigned long long)(const void*)&As[0][0][0];
  const unsigned BsBase = (unsigned)(unsigned long long)(const void*)&Bs[0][0][0];
  constexpr unsigned ABUF = BM * BK * 2, BBUF = BN * BK * 2;

  // ---- tile staging (async DMA for bf16 sides; VALU transpose for f32 B) ----
  auto stageA = [&](int k0, int buf) {
    for (int c = tid; c < BM * 4; c += 256) {           // 16B chunks, 4/row
      const int row = c >> 2, co = (c & 3) << 4;
      const unsigned lds = AsBase + (unsigned)buf * ABUF + (unsigned)(row * (BK * 2) + co);
      const unsigned long long g = (unsigned long long)(const void*)
          (A + (long long)(tileM + row) * lda + k0) + (unsigned)co;
      asm volatile("global_load_async_to_lds_b128 %0, %1, off"
                   :: "v"(lds), "v"(g) : "memory");
    }
  };
  auto stageB = [&](int k0, int buf) {
    if (BF16B) {
      for (int c = tid; c < BN * 4; c += 256) {
        const int row = c >> 2, co = (c & 3) << 4;
        const unsigned lds = BsBase + (unsigned)buf * BBUF + (unsigned)(row * (BK * 2) + co);
        const unsigned long long g = (unsigned long long)(const void*)
            (BH + (long long)(tileN + row) * ldb + k0) + (unsigned)co;
        asm volatile("global_load_async_to_lds_b128 %0, %1, off"
                     :: "v"(lds), "v"(g) : "memory");
      }
    } else {
      // f32 [K,N]: coalesced reads along N, transpose-convert into N-major LDS
      const int kq = (tid >> 5) << 2;     // 0,4,...,28
      const int nl = tid & 31;
      for (int nn = nl; nn < BN; nn += 32) {
        u16x4 p;
        p.x = f2bf(BF[(long long)(k0 + kq + 0) * ldb + (tileN + nn)]);
        p.y = f2bf(BF[(long long)(k0 + kq + 1) * ldb + (tileN + nn)]);
        p.z = f2bf(BF[(long long)(k0 + kq + 2) * ldb + (tileN + nn)]);
        p.w = f2bf(BF[(long long)(k0 + kq + 3) * ldb + (tileN + nn)]);
        *(u16x4*)&Bs[buf][nn][kq] = p;
      }
    }
  };

  f32x8 acc[FM][FN];
  for (int i = 0; i < FM; ++i)
    for (int j = 0; j < FN; ++j)
      for (int e = 0; e < 8; ++e) acc[i][j][e] = 0.0f;

  // ---- software pipeline: prime buffer 0 ----
  stageB(0, 0);
  stageA(0, 0);
  asm volatile("s_wait_asynccnt 0x0" ::: "memory");
  __syncthreads();

  int cur = 0;
  for (int k0 = 0; k0 < K; k0 += BK) {
    const bool more = (k0 + BK) < K;
    if (more) {                       // overlap next tile's DMA with WMMAs
      stageB(k0 + BK, cur ^ 1);
      stageA(k0 + BK, cur ^ 1);
    }

    // ---- B fragments: lanes 0-15 hold K 0..15, lanes 16-31 hold K 16..31
    Frag bfr[FN];
    const int kb_b = (lane >> 4) << 4;    // 0 or 16
    for (int fn = 0; fn < FN; ++fn) {
      const int n = wc * WTN + fn * 16 + (lane & 15);
      bfr[fn].h[0] = *(const u16x8*)&Bs[cur][n][kb_b];
      bfr[fn].h[1] = *(const u16x8*)&Bs[cur][n][kb_b + 8];
    }
    // ---- A fragments: lanes 0-15: K 0..15; lanes 16-31: K 16..31
    const int kb_a = (lane >> 4) << 4;    // 0 or 16
    for (int fm = 0; fm < FM; ++fm) {
      Frag afr;
      const int m = wr * WTM + fm * 16 + (lane & 15);
      afr.h[0] = *(const u16x8*)&As[cur][m][kb_a];
      afr.h[1] = *(const u16x8*)&As[cur][m][kb_a + 8];
      for (int fn = 0; fn < FN; ++fn) {
        acc[fm][fn] = __builtin_amdgcn_wmma_f32_16x16x32_bf16(
            false, afr.bf, false, bfr[fn].bf, (short)0, acc[fm][fn], false, false);
      }
    }

    if (more) {
      asm volatile("s_wait_asynccnt 0x0" ::: "memory");
      __syncthreads();
      cur ^= 1;
    }
  }

  // ---- epilogue: C 16x16 f32 layout: VGPR r -> M=r+8*(lane>=16), N=lane&15
  for (int fm = 0; fm < FM; ++fm) {
    const int mBase = tileM + wr * WTM + fm * 16 + ((lane >> 4) << 3);
    for (int fn = 0; fn < FN; ++fn) {
      const int col = tileN + wc * WTN + fn * 16 + (lane & 15);
      for (int r = 0; r < 8; ++r) {
        const int m = mBase + r;
        float v = acc[fm][fn][r] * alpha;
        if (HASBIAS) v += bias[col];
        if (HASADD)  v += Ad[(long long)m * ldadd + col];
        if (GELU)    v = 0.5f * v * (1.0f + erff(v * 0.70710678118654752f));
        if (HASRES)  v += R[(long long)m * ldc + col];
        if (OUTBF) CH[(long long)m * ldc + col] = f2bf(v);
        else       CF[(long long)m * ldc + col] = v;
      }
    }
  }
}

// =====================================================================
// Weight pre-pass: f32 [K,N] -> bf16 [N,K] (transpose + convert), tiled
// 32x32 through LDS so both global sides are coalesced. blockIdx.z = layer.
// =====================================================================
__global__ __launch_bounds__(256) void transpose_convert_bf16(
    const float* __restrict__ W, unsigned short* __restrict__ Wt, int K, int N)
{
  __shared__ float t[32][33];
  const long long zo = (long long)blockIdx.z * K * N;
  const float* Wz = W + zo;
  unsigned short* Wtz = Wt + zo;

  const int kb = blockIdx.y * 32, nb = blockIdx.x * 32;
  const int tx = threadIdx.x & 31, ty = threadIdx.x >> 5;   // 32x8
  for (int i = ty; i < 32; i += 8)
    t[i][tx] = Wz[(long long)(kb + i) * N + (nb + tx)];
  __syncthreads();
  for (int i = ty; i < 32; i += 8)
    Wtz[(long long)(nb + i) * K + (kb + tx)] = f2bf(t[tx][i]);
}

// =====================================================================
// Embedding + sinusoidal positional encoding (f32 residual stream)
// =====================================================================
__global__ __launch_bounds__(256) void embed_pe(const int* __restrict__ src,
                                                const float* __restrict__ emb,
                                                float* __restrict__ x)
{
  const long long idx = (long long)blockIdx.x * 256 + threadIdx.x;
  if (idx >= (long long)Mc * Dc) return;
  const int d = (int)(idx % Dc);
  const long long bl = idx / Dc;
  const int l = (int)(bl % Lc);
  const int tok = src[bl];
  const float div = __expf((float)(d & ~1) * (-9.210340371976184f / (float)Dc));
  const float ang = (float)l * div;
  const float pe = (d & 1) ? __cosf(ang) : __sinf(ang);
  x[idx] = emb[(long long)tok * Dc + d] * 32.0f + pe;   // sqrt(1024) = 32
}

// =====================================================================
// Special-token additive attention bias  [B,L,L]
// PAD=0, MASK=1, CLS0=2, CLS1=3
// =====================================================================
__global__ __launch_bounds__(256) void build_bias(const int* __restrict__ src,
                                                  float* __restrict__ bias)
{
  const long long idx = (long long)blockIdx.x * 256 + threadIdx.x;
  if (idx >= (long long)Bc * Lc * Lc) return;
  const int j = (int)(idx % Lc);
  const long long t = idx / Lc;
  const int i = (int)(t % Lc);
  const int b = (int)(t / Lc);
  const int ti = src[(long long)b * Lc + i];
  const int tj = src[(long long)b * Lc + j];
  const bool pi = (ti == 0), pj = (tj == 0);
  const bool ci = (ti >= 1) && (ti <= 3);
  const bool cj = (tj >= 1) && (tj <= 3);
  float m = 1.0f;
  if (cj && !pi && !ci)      m = 2.0f;   // col class, row normal
  if (ci && cj && (i != j))  m = 1.5f;   // both class, off-diagonal
  if (pi || pj)              m = 0.0f;   // pad zeroing wins
  bias[idx] = m;
}

// =====================================================================
// LayerNorm over last dim; output bf16 (consumed only as GEMM A operand)
// =====================================================================
__global__ __launch_bounds__(256) void layernorm_bf16(const float* __restrict__ x,
                                                      unsigned short* __restrict__ y,
                                                      const float* __restrict__ g,
                                                      const float* __restrict__ b,
                                                      int D)
{
  __shared__ float red[256];
  const int tid = threadIdx.x;
  const float*    xr = x + (long long)blockIdx.x * D;
  unsigned short* yr = y + (long long)blockIdx.x * D;

  float s = 0.0f;
  for (int i = tid; i < D; i += 256) s += xr[i];
  red[tid] = s; __syncthreads();
  for (int st = 128; st > 0; st >>= 1) {
    if (tid < st) red[tid] += red[tid + st];
    __syncthreads();
  }
  const float mean = red[0] / (float)D;
  __syncthreads();

  float v = 0.0f;
  for (int i = tid; i < D; i += 256) { const float d = xr[i] - mean; v += d * d; }
  red[tid] = v; __syncthreads();
  for (int st = 128; st > 0; st >>= 1) {
    if (tid < st) red[tid] += red[tid + st];
    __syncthreads();
  }
  const float inv = rsqrtf(red[0] / (float)D + 1e-5f);

  for (int i = tid; i < D; i += 256)
    yr[i] = f2bf((xr[i] - mean) * inv * g[i] + b[i]);
}

// =====================================================================
// Softmax over rows of exactly 512: 2 elems/thread held in registers,
// reads f32 scores once, writes bf16 probabilities only.
// =====================================================================
__global__ __launch_bounds__(256) void softmax512_bf16(const float* __restrict__ s,
                                                       unsigned short* __restrict__ o)
{
  __shared__ float red[256];
  const int tid = threadIdx.x;
  const float*    row  = s + (long long)blockIdx.x * 512;
  unsigned short* orow = o + (long long)blockIdx.x * 512;

  const float a = row[tid], b = row[tid + 256];
  red[tid] = fmaxf(a, b); __syncthreads();
  for (int st = 128; st > 0; st >>= 1) {
    if (tid < st) red[tid] = fmaxf(red[tid], red[tid + st]);
    __syncthreads();
  }
  const float mx = red[0]; __syncthreads();

  const float ea = __expf(a - mx), eb = __expf(b - mx);
  red[tid] = ea + eb; __syncthreads();
  for (int st = 128; st > 0; st >>= 1) {
    if (tid < st) red[tid] += red[tid + st];
    __syncthreads();
  }
  const float inv = 1.0f / red[0];
  orow[tid]       = f2bf(ea * inv);
  orow[tid + 256] = f2bf(eb * inv);
}

// =====================================================================
// Host orchestration
// =====================================================================
extern "C" void kernel_launch(void* const* d_in, const int* in_sizes, int n_in,
                              void* d_out, int out_size, void* d_ws, size_t ws_size,
                              hipStream_t stream) {
  (void)in_sizes; (void)n_in; (void)out_size; (void)ws_size;

  const int*   src   = (const int*)  d_in[0];
  const float* emb   = (const float*)d_in[1];
  const float* Wq    = (const float*)d_in[2];
  const float* bq    = (const float*)d_in[3];
  const float* Wk    = (const float*)d_in[4];
  const float* bk    = (const float*)d_in[5];
  const float* Wv    = (const float*)d_in[6];
  const float* bv    = (const float*)d_in[7];
  const float* Wo    = (const float*)d_in[8];
  const float* bo    = (const float*)d_in[9];
  const float* ln1g  = (const float*)d_in[10];
  const float* ln1b  = (const float*)d_in[11];
  const float* ln2g  = (const float*)d_in[12];
  const float* ln2b  = (const float*)d_in[13];
  const float* W1    = (const float*)d_in[14];
  const float* b1    = (const float*)d_in[15];
  const float* W2    = (const float*)d_in[16];
  const float* b2    = (const float*)d_in[17];
  const float* lnfg  = (const float*)d_in[18];
  const float* lnfb  = (const float*)d_in[19];
  const float* Wout  = (const float*)d_in[20];
  const float* bout  = (const float*)d_in[21];
  float* out = (float*)d_out;

  // -------- workspace layout: f32 region then bf16 (ushort) region --------
  const long long MD   = (long long)Mc * Dc;
  const long long MF   = (long long)Mc * Fc;
  const long long LL   = (long long)Lc * Lc;
  const long long LD   = (long long)Lc * Dc;
  const long long BLL  = (long long)Bc * LL;
  const long long BHLL = (long long)Bc * Hc * LL;

  float* ws     = (float*)d_ws;
  float* x      = ws;                // [M,D] f32 residual stream
  float* vbuf   = x + MD;            // [M,D] f32 V (transposed in A.V B-stage)
  float* scores = vbuf + MD;         // [B*H,L,L] f32 pre-softmax scores
  float* biasM  = scores + BHLL;     // [B,L,L]

  unsigned short* hbf   = (unsigned short*)(biasM + BLL); // [M,D] LN out / attn out
  unsigned short* qbf   = hbf + MD;                       // [M,D]
  unsigned short* kbf   = qbf + MD;                       // [M,D]
  unsigned short* ff1bf = kbf + MD;                       // [M,FF]
  unsigned short* scbf  = ff1bf + MF;                     // [B*H,L,L] probs
  const long long WDD = (long long)NLc * Dc * Dc;
  const long long WDF = (long long)NLc * Dc * Fc;
  unsigned short* Wqt   = scbf + BHLL;                    // [6][D][D] (N-major)
  unsigned short* Wkt   = Wqt + WDD;
  unsigned short* Wvt   = Wkt + WDD;
  unsigned short* Wot   = Wvt + WDD;
  unsigned short* W1t   = Wot + WDD;                      // [6][FF][D]
  unsigned short* W2t   = W1t + WDF;                      // [6][D][FF]
  unsigned short* Woutt = W2t + WDF;                      // [V][D]

  // -------- weight pre-pass: transpose + convert to bf16 --------
  transpose_convert_bf16<<<dim3(Dc / 32, Dc / 32, NLc), 256, 0, stream>>>(Wq, Wqt, Dc, Dc);
  transpose_convert_bf16<<<dim3(Dc / 32, Dc / 32, NLc), 256, 0, stream>>>(Wk, Wkt, Dc, Dc);
  transpose_convert_bf16<<<dim3(Dc / 32, Dc / 32, NLc), 256, 0, stream>>>(Wv, Wvt, Dc, Dc);
  transpose_convert_bf16<<<dim3(Dc / 32, Dc / 32, NLc), 256, 0, stream>>>(Wo, Wot, Dc, Dc);
  transpose_convert_bf16<<<dim3(Fc / 32, Dc / 32, NLc), 256, 0, stream>>>(W1, W1t, Dc, Fc);
  transpose_convert_bf16<<<dim3(Dc / 32, Fc / 32, NLc), 256, 0, stream>>>(W2, W2t, Fc, Dc);
  transpose_convert_bf16<<<dim3(Vc / 32, Dc / 32, 1),   256, 0, stream>>>(Wout, Woutt, Dc, Vc);

  // -------- prologue --------
  build_bias<<<(int)((BLL + 255) / 256), 256, 0, stream>>>(src, biasM);
  embed_pe<<<(int)((MD + 255) / 256), 256, 0, stream>>>(src, emb, x);

  const dim3 gMain(Dc / 128, Mc / 128, 1);
  const dim3 gFF1(Fc / 128, Mc / 128, 1);
  const dim3 gScores(Lc / 128, Lc / 128, Bc * Hc);
  const dim3 gAV(1, Lc / 128, Bc * Hc);
  const dim3 gLogit(Vc / 128, Mc / 128, 1);

  for (int l = 0; l < NLc; ++l) {
    const long long lDD = (long long)l * Dc * Dc;
    const long long lDF = (long long)l * Dc * Fc;
    const long long lD  = (long long)l * Dc;
    // --- LN1 -> bf16
    layernorm_bf16<<<Mc, 256, 0, stream>>>(x, hbf, ln1g + lD, ln1b + lD, Dc);
    // --- Q, K (bf16 out), V (f32 out, transposed later in A.V B-stage)
    gemm_wmma<128,128,2,4,true,true,false,true,false,false><<<gMain, 256, 0, stream>>>(
        hbf, Wqt + lDD, qbf, bq + lD, nullptr, nullptr,
        Dc, Dc, Dc, Dc, 0, 0,0, 0,0, 0,0, 0, 1, 1.0f);
    gemm_wmma<128,128,2,4,true,true,false,true,false,false><<<gMain, 256, 0, stream>>>(
        hbf, Wkt + lDD, kbf, bk + lD, nullptr, nullptr,
        Dc, Dc, Dc, Dc, 0, 0,0, 0,0, 0,0, 0, 1, 1.0f);
    gemm_wmma<128,128,2,4,true,false,false,true,false,false><<<gMain, 256, 0, stream>>>(
        hbf, Wvt + lDD, vbuf, bv + lD, nullptr, nullptr,
        Dc, Dc, Dc, Dc, 0, 0,0, 0,0, 0,0, 0, 1, 1.0f);
    // --- scores = 0.125 * Q @ K^T + biasM  (both operands bf16, async-staged)
    gemm_wmma<128,128,2,4,true,false,false,false,false,true><<<gScores, 256, 0, stream>>>(
        qbf, kbf, scores, nullptr, nullptr, biasM,
        DHc, Dc, Dc, Lc, Lc,
        LD, DHc,                      // A: b*L*D + h*DH
        LD, DHc,                      // B: same
        (long long)Hc * LL, LL,       // C: batch*L*L
        LL,                           // add: per-b, broadcast over heads
        Hc, 0.125f);
    // --- softmax -> bf16 probs
    softmax512_bf16<<<Bc * Hc * Lc, 256, 0, stream>>>(scores, scbf);
    // --- attn = P @ V  (P bf16 async; V f32 transpose-stage) -> bf16 hbf
    gemm_wmma<128,64,4,2,false,true,false,false,false,false><<<gAV, 256, 0, stream>>>(
        scbf, vbuf, hbf, nullptr, nullptr, nullptr,
        Lc, Lc, Dc, Dc, 0,
        (long long)Hc * LL, LL,       // A: batch*L*L
        LD, DHc,                      // B (V): b*L*D + h*DH
        LD, DHc,                      // C: same column slice
        0, Hc, 1.0f);
    // --- x = x + attn @ Wo + bo   (f32 residual out)
    gemm_wmma<128,128,2,4,true,false,false,true,true,false><<<gMain, 256, 0, stream>>>(
        hbf, Wot + lDD, x, bo + lD, x, nullptr,
        Dc, Dc, Dc, Dc, 0, 0,0, 0,0, 0,0, 0, 1, 1.0f);
    // --- LN2 -> bf16
    layernorm_bf16<<<Mc, 256, 0, stream>>>(x, hbf, ln2g + lD, ln2b + lD, Dc);
    // --- ff1 = gelu(h @ W1 + b1) -> bf16
    gemm_wmma<128,128,2,4,true,true,true,true,false,false><<<gFF1, 256, 0, stream>>>(
        hbf, W1t + lDF, ff1bf, b1 + (long long)l * Fc, nullptr, nullptr,
        Dc, Dc, Dc, Fc, 0, 0,0, 0,0, 0,0, 0, 1, 1.0f);
    // --- x = x + ff1 @ W2 + b2   (f32 residual out)
    gemm_wmma<128,128,2,4,true,false,false,true,true,false><<<gMain, 256, 0, stream>>>(
        ff1bf, W2t + lDF, x, b2 + lD, x, nullptr,
        Fc, Fc, Fc, Dc, 0, 0,0, 0,0, 0,0, 0, 1, 1.0f);
  }

  // final LN + logits (f32 out to d_out)
  layernorm_bf16<<<Mc, 256, 0, stream>>>(x, hbf, lnfg, lnfb, Dc);
  gemm_wmma<128,128,2,4,true,false,false,true,false,false><<<gLogit, 256, 0, stream>>>(
      hbf, Woutt, out, bout, nullptr, nullptr,
      Dc, Dc, Dc, Vc, 0, 0,0, 0,0, 0,0, 0, 1, 1.0f);
}
